// InvariantSlotAttention_69939247448162
// MI455X (gfx1250) — matrix operand order repaired and checked by reading, not compiled, not run on hardware
//
#include <hip/hip_runtime.h>
#include <hip/hip_bf16.h>

typedef __attribute__((ext_vector_type(16))) __bf16 v16bf;
typedef __attribute__((ext_vector_type(8)))  __bf16 v8bf;
typedef __attribute__((ext_vector_type(8)))  float  v8f;
typedef __attribute__((ext_vector_type(4)))  float  v4f;

#define NB 16
#define NN 4096
#define ND 64
#define NK 8

// 16-bit WMMA fragment mapping (ISA 7.12.2), per lane:
//   element e in [0,8)  <-> K = e + 8*hi
//   element e in [8,16) <-> K = (e-8) + 16 + 8*hi
// i.e. two contiguous 8-element runs. Inverse (K -> hi,e):
//   hi = (K>>3)&1 ; tt = K & ~8 ; e = (tt<8) ? tt : tt-8
__device__ __forceinline__ void k_to_he(int k32, int& hi, int& e) {
  hi = (k32 >> 3) & 1;
  const int tt = k32 & 0x17;
  e = (tt < 8) ? tt : (tt - 8);
}

// ---------------------------------------------------------------------------
// Kernel 0: copy initial slot state into workspace
// ---------------------------------------------------------------------------
__global__ __launch_bounds__(256) void init_state_kernel(
    const float* __restrict__ s0, const float* __restrict__ p0,
    const float* __restrict__ c0, float* __restrict__ slots,
    float* __restrict__ pos, float* __restrict__ sc) {
  const int t = blockIdx.x * blockDim.x + threadIdx.x;
  const int stride = gridDim.x * blockDim.x;
  for (int i = t; i < NB * NK * ND; i += stride) slots[i] = s0[i];
  for (int i = t; i < NB * NK * 2; i += stride) { pos[i] = p0[i]; sc[i] = c0[i]; }
}

// ---------------------------------------------------------------------------
// Kernel 1: fused LayerNorm + k/v projection.
//  k -> row-major bf16 [B*N, D] (consumed as A-fragments: contiguous b128s)
//  v -> B-fragment-major bf16   (consumed as B-fragments: contiguous b128s)
// Weights staged in LDS pre-swizzled to fragment order; each wave owns one
// (mat, ct) pair and keeps its B-fragments in registers across 4 row tiles.
// ---------------------------------------------------------------------------
__global__ __launch_bounds__(256) void kv_proj_kernel(
    const float* __restrict__ inp, const float* __restrict__ lng,
    const float* __restrict__ lnb, const float* __restrict__ Wk,
    const float* __restrict__ Wv, __bf16* __restrict__ kout,
    __bf16* __restrict__ vfrag) {
  __shared__ __align__(32) __bf16 sX[64][64];                 // 8 KB
  __shared__ __align__(32) __bf16 sWf[2 * 2 * 4 * 32 * 16];   // 16 KB frag-major
  const int tid = threadIdx.x, lane = tid & 31, wave = tid >> 5;
  const long rowbase = (long)blockIdx.x * 64;

  // Stage Wk/Wv in fragment-major order: [mat][c][ct][lane][e]
  for (int i = tid; i < 2 * 64 * 64; i += 256) {
    const int mat = i >> 12;
    const int idx = i & 4095;
    const int din = idx >> 6, dout = idx & 63;
    const float w = mat ? Wv[idx] : Wk[idx];
    const int c = din >> 5;
    int hi, e;
    k_to_he(din & 31, hi, e);
    const int lanef = hi * 16 + (dout & 15);
    const int ct = dout >> 4;
    sWf[(((mat * 2 + c) * 4 + ct) * 32 + lanef) * 16 + e] = (__bf16)w;
  }

  // LayerNorm 8 rows per wave (wave32 shuffle reductions)
  const float g0 = lng[lane], g1 = lng[lane + 32];
  const float b0 = lnb[lane], b1 = lnb[lane + 32];
  for (int i = 0; i < 8; ++i) {
    const int r = wave * 8 + i;
    const float* row = inp + (rowbase + r) * ND;
    const float x0 = row[lane], x1 = row[lane + 32];
    float s = x0 + x1;
#pragma unroll
    for (int m = 16; m >= 1; m >>= 1) s += __shfl_xor(s, m, 32);
    const float mean = s * (1.0f / 64.0f);
    const float d0 = x0 - mean, d1 = x1 - mean;
    float vs = d0 * d0 + d1 * d1;
#pragma unroll
    for (int m = 16; m >= 1; m >>= 1) vs += __shfl_xor(vs, m, 32);
    const float rstd = rsqrtf(vs * (1.0f / 64.0f) + 1e-5f);
    sX[r][lane]      = (__bf16)(d0 * rstd * g0 + b0);
    sX[r][lane + 32] = (__bf16)(d1 * rstd * g1 + b1);
  }
  __syncthreads();

  const int hi = lane >> 4;
  const int mn = lane & 15;
  const int mat = wave >> 2, ct = wave & 3;
  // B-fragments: one contiguous 32B read each, kept in registers for all rt
  const v16bf bf0 = *(const v16bf*)&sWf[(((mat * 2 + 0) * 4 + ct) * 32 + lane) * 16];
  const v16bf bf1 = *(const v16bf*)&sWf[(((mat * 2 + 1) * 4 + ct) * 32 + lane) * 16];

  for (int rt = 0; rt < 4; ++rt) {
    v8f acc;
#pragma unroll
    for (int r = 0; r < 8; ++r) acc[r] = 0.0f;
#pragma unroll
    for (int c = 0; c < 2; ++c) {
      // A-fragment: two contiguous b128 LDS loads per lane
      const __bf16* xr = &sX[rt * 16 + mn][c * 32 + hi * 8];
      const v8bf alo = *(const v8bf*)xr;
      const v8bf ahi = *(const v8bf*)(xr + 16);
      const v16bf af = __builtin_shufflevector(alo, ahi, 0, 1, 2, 3, 4, 5, 6, 7,
                                               8, 9, 10, 11, 12, 13, 14, 15);
      acc = __builtin_amdgcn_wmma_f32_16x16x32_bf16(false, af, false,
                                                    c ? bf1 : bf0,
                                                    (short)0, acc, false, false);
    }
    if (mat == 0) {
#pragma unroll
      for (int r = 0; r < 8; ++r) {
        const long rowg = rowbase + rt * 16 + r + hi * 8;
        kout[rowg * ND + ct * 16 + mn] = (__bf16)acc[r];
      }
    } else {
      // scatter v into B-fragment-major order: [b][nchunk32][dt=ct][lane][e]
#pragma unroll
      for (int r = 0; r < 8; ++r) {
        const long rowg = rowbase + rt * 16 + r + hi * 8;
        const int bb = (int)(rowg >> 12);
        const int n = (int)(rowg & 4095);
        const int nchunk = n >> 5;
        int hv, e;
        k_to_he(n & 31, hv, e);
        vfrag[((((long)bb * 128 + nchunk) * 4 + ct) * 32 + hv * 16 + mn) * 16 + e] =
            (__bf16)acc[r];
      }
    }
  }
}

// ---------------------------------------------------------------------------
// Kernel 2 (per iter): q = LN(slots)@Wq * D^-0.5 + per-slot rel-key dots;
// zeros the moment / raw-update accumulators.
// ---------------------------------------------------------------------------
__global__ __launch_bounds__(128) void q_prep_kernel(
    const float* __restrict__ slots, const float* __restrict__ lng,
    const float* __restrict__ lnb, const float* __restrict__ Wq,
    const float* __restrict__ Wg_k, float* __restrict__ qout,
    float* __restrict__ about, float* __restrict__ moments,
    float* __restrict__ updraw) {
  __shared__ float sx[128][ND];
  const int t = threadIdx.x;
  for (int i = t; i < NB * NK * ND; i += 128) updraw[i] = 0.0f;
  for (int i = t; i < NB * NK * 8; i += 128) moments[i] = 0.0f;
  const float* row = slots + t * ND;
  float mean = 0.f;
  for (int d = 0; d < ND; ++d) mean += row[d];
  mean *= (1.0f / 64.0f);
  float var = 0.f;
  for (int d = 0; d < ND; ++d) { const float x = row[d] - mean; sx[t][d] = x; var += x * x; }
  const float rstd = rsqrtf(var * (1.0f / 64.0f) + 1e-5f);
  for (int d = 0; d < ND; ++d) sx[t][d] = sx[t][d] * rstd * lng[d] + lnb[d];
  float as = 0.f, bs = 0.f;
  for (int j = 0; j < ND; ++j) {
    float q = 0.f;
    for (int e = 0; e < ND; ++e) q += sx[t][e] * Wq[e * ND + j];
    q *= 0.125f;  // 64^-0.5
    qout[t * ND + j] = q;
    as += q * Wg_k[j];
    bs += q * Wg_k[ND + j];
  }
  about[t * 2] = as;
  about[t * 2 + 1] = bs;
}

// ---------------------------------------------------------------------------
// Kernel 3 (per iter): logits via WMMA ([n x slot] tiles), rank-2 rel term,
// softmax over slots across 8-lane groups, vectorized b128 attn stores.
// ---------------------------------------------------------------------------
__global__ __launch_bounds__(256) void attn_kernel(
    const __bf16* __restrict__ kbuf, const float* __restrict__ qbuf,
    const float* __restrict__ ab, const float* __restrict__ grid,
    const float* __restrict__ pos, const float* __restrict__ sc,
    float* __restrict__ attn) {
  __shared__ float gx[512], gy[512];
  const int tid = threadIdx.x, lane = tid & 31, wave = tid >> 5;
  const int b = blockIdx.x >> 3;
  const int nbase = (blockIdx.x & 7) * 512;
  for (int i = tid; i < 512; i += 256) {
    gx[i] = grid[(nbase + i) * 2];
    gy[i] = grid[(nbase + i) * 2 + 1];
  }
  __syncthreads();
  const int hi = lane >> 4;
  const int mn = lane & 15;
  const bool sval = (mn < NK);
  v16bf qf[2];
#pragma unroll
  for (int e = 0; e < 16; ++e) { qf[0][e] = (__bf16)0.0f; qf[1][e] = (__bf16)0.0f; }
  float as = 0.f, bs = 0.f, pxs = 0.f, pys = 0.f, rsx = 0.f, rsy = 0.f;
  if (sval) {
    const int sk = b * NK + mn;
    const float* q = qbuf + sk * ND;
#pragma unroll
    for (int e = 0; e < 16; ++e) {
      const int d = (e < 8) ? (e + 8 * hi) : (e + 8 + 8 * hi);
      qf[0][e] = (__bf16)q[d];
      qf[1][e] = (__bf16)q[d + 32];
    }
    as = ab[sk * 2];
    bs = ab[sk * 2 + 1];
    pxs = pos[sk * 2];
    pys = pos[sk * 2 + 1];
    rsx = 1.0f / (sc[sk * 2] + 1e-8f);
    rsy = 1.0f / (sc[sk * 2 + 1] + 1e-8f);
  }
  for (int t4 = 0; t4 < 4; ++t4) {
    const int ntl = wave * 64 + t4 * 16;
    const long rowb = (long)b * NN + nbase + ntl;
    if (t4 < 3)  // prefetch next tile's k rows (speculative, gfx1250 path)
      __builtin_prefetch((const void*)(kbuf + (rowb + 16 + mn) * ND), 0, 0);
    v8f acc;
#pragma unroll
    for (int r = 0; r < 8; ++r) acc[r] = 0.0f;
#pragma unroll
    for (int c = 0; c < 2; ++c) {
      // A-fragment: two contiguous b128 global loads
      const __bf16* kr = kbuf + (rowb + mn) * ND + c * 32 + hi * 8;
      const v8bf alo = *(const v8bf*)kr;
      const v8bf ahi = *(const v8bf*)(kr + 16);
      const v16bf af = __builtin_shufflevector(alo, ahi, 0, 1, 2, 3, 4, 5, 6, 7,
                                               8, 9, 10, 11, 12, 13, 14, 15);
      acc = __builtin_amdgcn_wmma_f32_16x16x32_bf16(false, af, false, qf[c],
                                                    (short)0, acc, false, false);
    }
    float vals[8];
#pragma unroll
    for (int r = 0; r < 8; ++r) {
      const int nl = ntl + r + hi * 8;
      const float relx = (gx[nl] - pxs) * rsx;
      const float rely = (gy[nl] - pys) * rsy;
      float lg = acc[r] + relx * as + rely * bs;
      if (!sval) lg = -1e30f;
      float mx = lg;
      mx = fmaxf(mx, __shfl_xor(mx, 1, 32));
      mx = fmaxf(mx, __shfl_xor(mx, 2, 32));
      mx = fmaxf(mx, __shfl_xor(mx, 4, 32));
      const float ev = __expf(lg - mx);
      float ssum = ev;
      ssum += __shfl_xor(ssum, 1, 32);
      ssum += __shfl_xor(ssum, 2, 32);
      ssum += __shfl_xor(ssum, 4, 32);
      vals[r] = ev / ssum;
    }
    if (sval) {  // two b128 stores per lane (8 consecutive n)
      float* op = attn + (long)(b * NK + mn) * NN + nbase + ntl + hi * 8;
      v4f s0, s1;
#pragma unroll
      for (int j = 0; j < 4; ++j) { s0[j] = vals[j]; s1[j] = vals[4 + j]; }
      *(v4f*)op = s0;
      *(v4f*)(op + 4) = s1;
    }
  }
}

// ---------------------------------------------------------------------------
// Kernel 4 (per iter): raw updates = attn @ v (WMMA f32 accum) + grid moments.
// attn read as b128 f32 vectors; v read as contiguous fragment-major b128s.
// ---------------------------------------------------------------------------
__global__ __launch_bounds__(256) void update_kernel(
    const __bf16* __restrict__ vfrag, const float* __restrict__ attn,
    const float* __restrict__ grid, float* __restrict__ updraw,
    float* __restrict__ moments) {
  __shared__ float gx[512], gy[512];
  __shared__ float lacc[NK * ND];
  __shared__ float lmom[NK * 8];
  const int tid = threadIdx.x, lane = tid & 31, wave = tid >> 5;
  const int b = blockIdx.x >> 3;
  const int nbase = (blockIdx.x & 7) * 512;
  for (int i = tid; i < 512; i += 256) {
    gx[i] = grid[(nbase + i) * 2];
    gy[i] = grid[(nbase + i) * 2 + 1];
    if (i < NK * ND) lacc[i] = 0.0f;
    if (i < NK * 8) lmom[i] = 0.0f;
  }
  __syncthreads();
  const int hi = lane >> 4;
  const int mn = lane & 15;
  const bool sval = (mn < NK);
  v8f acc[4];
#pragma unroll
  for (int dt = 0; dt < 4; ++dt)
#pragma unroll
    for (int r = 0; r < 8; ++r) acc[dt][r] = 0.0f;
  float m0 = 0.f, m1 = 0.f, m2 = 0.f, m3 = 0.f, m4 = 0.f;
#pragma unroll
  for (int c = 0; c < 2; ++c) {
    const int nb2 = wave * 64 + c * 32;
    const long fbase = (((long)b * 128 + ((nbase + nb2) >> 5)) * 4) * 32 * 16;
    if (c == 0)  // prefetch second chunk's v fragments
      __builtin_prefetch((const void*)&vfrag[(((long)b * 128 + ((nbase + 32) >> 5) + wave * 2) * 4) * 32 * 16],
                         0, 0);
    // A-fragment (attn, f32 -> bf16) via 4 b128 loads; f32 kept for moments
    float fv[16];
#pragma unroll
    for (int e = 0; e < 16; ++e) fv[e] = 0.0f;
    if (sval) {
      const float* ap = attn + (long)(b * NK + mn) * NN + nbase + nb2 + hi * 8;
      const v4f a0 = *(const v4f*)(ap);
      const v4f a1 = *(const v4f*)(ap + 4);
      const v4f a2 = *(const v4f*)(ap + 16);
      const v4f a3 = *(const v4f*)(ap + 20);
#pragma unroll
      for (int j = 0; j < 4; ++j) {
        fv[j] = a0[j]; fv[4 + j] = a1[j]; fv[8 + j] = a2[j]; fv[12 + j] = a3[j];
      }
      const int n1 = nb2 + hi * 8, n2 = nb2 + 16 + hi * 8;
#pragma unroll
      for (int j = 0; j < 8; ++j) {
        const float a = fv[j], x = gx[n1 + j], y = gy[n1 + j];
        m0 += a; m1 += a * x; m2 += a * y; m3 += a * x * x; m4 += a * y * y;
        const float a2v = fv[8 + j], x2 = gx[n2 + j], y2 = gy[n2 + j];
        m0 += a2v; m1 += a2v * x2; m2 += a2v * y2; m3 += a2v * x2 * x2; m4 += a2v * y2 * y2;
      }
    }
    v16bf af;
#pragma unroll
    for (int e = 0; e < 16; ++e) af[e] = (__bf16)fv[e];
#pragma unroll
    for (int dt = 0; dt < 4; ++dt) {
      const v16bf bf = *(const v16bf*)&vfrag[fbase + ((long)dt * 32 + lane) * 16];
      acc[dt] = __builtin_amdgcn_wmma_f32_16x16x32_bf16(false, af, false, bf,
                                                        (short)0, acc[dt], false, false);
    }
  }
  m0 += __shfl_xor(m0, 16, 32);
  m1 += __shfl_xor(m1, 16, 32);
  m2 += __shfl_xor(m2, 16, 32);
  m3 += __shfl_xor(m3, 16, 32);
  m4 += __shfl_xor(m4, 16, 32);
  if (lane < NK) {
    atomicAdd(&lmom[lane * 8 + 0], m0);
    atomicAdd(&lmom[lane * 8 + 1], m1);
    atomicAdd(&lmom[lane * 8 + 2], m2);
    atomicAdd(&lmom[lane * 8 + 3], m3);
    atomicAdd(&lmom[lane * 8 + 4], m4);
  }
  if (!hi) {
#pragma unroll
    for (int dt = 0; dt < 4; ++dt)
#pragma unroll
      for (int r = 0; r < 8; ++r)
        atomicAdd(&lacc[r * ND + dt * 16 + mn], acc[dt][r]);
  }
  __syncthreads();
  for (int i = tid; i < NK * ND; i += 256) atomicAdd(&updraw[b * NK * ND + i], lacc[i]);
  for (int i = tid; i < NK * 8; i += 256) atomicAdd(&moments[b * NK * 8 + i], lmom[i]);
}

// ---------------------------------------------------------------------------
// Kernel 5 (per iter): finalize updates/pos/sc from moments, GRU cell,
// residual LN-MLP. One block, thread = (b,k) row.
// ---------------------------------------------------------------------------
__global__ __launch_bounds__(128) void slot_update_kernel(
    const float* __restrict__ moments, const float* __restrict__ updraw,
    const float* __restrict__ Wg_v, float* __restrict__ slots,
    float* __restrict__ pos, float* __restrict__ sc,
    const float* __restrict__ wih, const float* __restrict__ whh,
    const float* __restrict__ bih, const float* __restrict__ bhh,
    const float* __restrict__ lng, const float* __restrict__ lnb,
    const float* __restrict__ w1, const float* __restrict__ b1,
    const float* __restrict__ w2, const float* __restrict__ b2,
    float* __restrict__ out, const int last) {
  __shared__ float sU[128][ND];
  __shared__ float sH[128][ND];
  const int t = threadIdx.x;
  const float* mom = moments + t * 8;
  const float S = mom[0], Sx = mom[1], Sy = mom[2], Sxx = mom[3], Syy = mom[4];
  const float inv = 1.0f / (S + 1e-8f);
  const float pxo = pos[t * 2], pyo = pos[t * 2 + 1];
  const float Srx = (Sx - pxo * S) / (sc[t * 2] + 1e-8f);
  const float Sry = (Sy - pyo * S) / (sc[t * 2 + 1] + 1e-8f);
  for (int d = 0; d < ND; ++d)
    sU[t][d] = (updraw[t * ND + d] + Srx * Wg_v[d] + Sry * Wg_v[ND + d]) * inv;
  const float px = Sx * inv, py = Sy * inv;
  const float vx = (Sxx - 2.f * px * Sx + px * px * S) * inv;
  const float vy = (Syy - 2.f * py * Sy + py * py * S) * inv;
  pos[t * 2] = px;
  pos[t * 2 + 1] = py;
  sc[t * 2] = sqrtf(vx + 1e-8f);
  sc[t * 2 + 1] = sqrtf(vy + 1e-8f);
  for (int d = 0; d < ND; ++d) sH[t][d] = slots[t * ND + d];
  float nh[ND];
  for (int d = 0; d < ND; ++d) {
    float ir = bih[d], iz = bih[ND + d], ig = bih[2 * ND + d];
    float hr = bhh[d], hz = bhh[ND + d], hg = bhh[2 * ND + d];
    for (int e = 0; e < ND; ++e) {
      const float u = sU[t][e], hh = sH[t][e];
      const float* wi = wih + e * 3 * ND + d;
      const float* wh = whh + e * 3 * ND + d;
      ir += u * wi[0];
      iz += u * wi[ND];
      ig += u * wi[2 * ND];
      hr += hh * wh[0];
      hz += hh * wh[ND];
      hg += hh * wh[2 * ND];
    }
    const float r = 1.0f / (1.0f + __expf(-(ir + hr)));
    const float z = 1.0f / (1.0f + __expf(-(iz + hz)));
    const float n = tanhf(ig + r * hg);
    nh[d] = (1.0f - z) * n + z * sH[t][d];
  }
  float mean = 0.f;
  for (int d = 0; d < ND; ++d) mean += nh[d];
  mean *= (1.0f / 64.0f);
  float var = 0.f;
  for (int d = 0; d < ND; ++d) { const float x = nh[d] - mean; var += x * x; }
  const float rstd = rsqrtf(var * (1.0f / 64.0f) + 1e-5f);
  for (int d = 0; d < ND; ++d) sH[t][d] = (nh[d] - mean) * rstd * lng[d] + lnb[d];
#pragma unroll 8
  for (int d = 0; d < ND; ++d) nh[d] += b2[d];
  for (int j = 0; j < 128; ++j) {
    float hv = b1[j];
    for (int e = 0; e < ND; ++e) hv += sH[t][e] * w1[e * 128 + j];
    hv = fmaxf(hv, 0.0f);
#pragma unroll 8
    for (int d = 0; d < ND; ++d) nh[d] += hv * w2[j * ND + d];
  }
  for (int d = 0; d < ND; ++d) {
    slots[t * ND + d] = nh[d];
    if (last) out[t * ND + d] = nh[d];
  }
}

// ---------------------------------------------------------------------------
extern "C" void kernel_launch(void* const* d_in, const int* in_sizes, int n_in,
                              void* d_out, int out_size, void* d_ws, size_t ws_size,
                              hipStream_t stream) {
  const float* inputs  = (const float*)d_in[0];
  const float* grid    = (const float*)d_in[1];
  const float* slots0  = (const float*)d_in[2];
  const float* pos0    = (const float*)d_in[3];
  const float* scale0  = (const float*)d_in[4];
  const float* Wq      = (const float*)d_in[5];
  const float* Wk      = (const float*)d_in[6];
  const float* Wv      = (const float*)d_in[7];
  const float* Wg_k    = (const float*)d_in[8];
  const float* Wg_v    = (const float*)d_in[9];
  const float* ln_in_g = (const float*)d_in[10];
  const float* ln_in_b = (const float*)d_in[11];
  const float* ln_s_g  = (const float*)d_in[12];
  const float* ln_s_b  = (const float*)d_in[13];
  const float* ln_m_g  = (const float*)d_in[14];
  const float* ln_m_b  = (const float*)d_in[15];
  const float* gwih    = (const float*)d_in[16];
  const float* gwhh    = (const float*)d_in[17];
  const float* gbih    = (const float*)d_in[18];
  const float* gbhh    = (const float*)d_in[19];
  const float* w1      = (const float*)d_in[20];
  const float* b1      = (const float*)d_in[21];
  const float* w2      = (const float*)d_in[22];
  const float* b2      = (const float*)d_in[23];

  char* ws = (char*)d_ws;
  size_t off = 0;
  __bf16* kbf   = (__bf16*)(ws + off); off += (size_t)NB * NN * ND * 2;  // 8 MB
  __bf16* vfrag = (__bf16*)(ws + off); off += (size_t)NB * NN * ND * 2;  // 8 MB
  float* attn   = (float*)(ws + off); off += (size_t)NB * NK * NN * 4;   // 2 MB
  float* qbuf   = (float*)(ws + off); off += (size_t)NB * NK * ND * 4;
  float* ab     = (float*)(ws + off); off += (size_t)NB * NK * 2 * 4;
  float* mom    = (float*)(ws + off); off += (size_t)NB * NK * 8 * 4;
  float* updraw = (float*)(ws + off); off += (size_t)NB * NK * ND * 4;
  float* slots  = (float*)(ws + off); off += (size_t)NB * NK * ND * 4;
  float* pos    = (float*)(ws + off); off += (size_t)NB * NK * 2 * 4;
  float* sc     = (float*)(ws + off); off += (size_t)NB * NK * 2 * 4;

  init_state_kernel<<<8, 256, 0, stream>>>(slots0, pos0, scale0, slots, pos, sc);
  kv_proj_kernel<<<(NB * NN) / 64, 256, 0, stream>>>(inputs, ln_in_g, ln_in_b,
                                                     Wk, Wv, kbf, vfrag);
  for (int it = 0; it < 3; ++it) {
    q_prep_kernel<<<1, 128, 0, stream>>>(slots, ln_s_g, ln_s_b, Wq, Wg_k,
                                         qbuf, ab, mom, updraw);
    attn_kernel<<<NB * 8, 256, 0, stream>>>(kbf, qbuf, ab, grid, pos, sc, attn);
    update_kernel<<<NB * 8, 256, 0, stream>>>(vfrag, attn, grid, updraw, mom);
    slot_update_kernel<<<1, 128, 0, stream>>>(mom, updraw, Wg_v, slots, pos, sc,
                                              gwih, gwhh, gbih, gbhh,
                                              ln_m_g, ln_m_b, w1, b1, w2, b2,
                                              (float*)d_out, (it == 2) ? 1 : 0);
  }
}
